// CriticReadoutModule_73959336837233
// MI455X (gfx1250) — compile-verified, matrix operand order checked by reading, not run on hardware
//
#include <hip/hip_runtime.h>

// ---------------------------------------------------------------------------
// CriticReadoutModule fused bf16-WMMA implementation for gfx1250 (MI455X).
//  - prep kernel: f32 -> bf16 weight conversion + WMMA B-fragment swizzle
//  - main kernel: per 16-row batch tile x twin: phi MLP per object (WMMA),
//    object sum-pool in registers, fused rho MLP (WMMA) -> Q scalar.
//  - ctx staging is software-pipelined with float4 loads: next object's
//    slice is loaded (b128) while the current object's GEMMs run; commits
//    to LDS as packed bf16 pairs (ds_store_b64).
// ---------------------------------------------------------------------------

typedef __attribute__((ext_vector_type(16))) __bf16 v16bf;
typedef __attribute__((ext_vector_type(8)))  float  v8f;

#define BATCH      32768
#define NB_OBJ     10
#define DIM_CTX    184
#define K1_REAL    264   // phi layer1 K (64 body + 16 act + 184 ctx)
#define K1_PAD     288   // padded to multiple of 32 (9 K-tiles)
#define N1         256   // phi hidden
#define K2         256   // phi layer2 K (8 K-tiles)
#define N2         192   // phi out (12 N-tiles)
#define KR         192   // rho layer1 K (6 K-tiles)
#define NR         256   // rho hidden (16 N-tiles)
#define XS_STRIDE  296   // LDS row stride (elems), multiple of 8 (16B align)
#define HS_STRIDE  264   // LDS row stride for h / p, multiple of 8
#define CTX_F4     46    // float4 chunks per ctx row (184/4)
#define CTX_CHUNKS 736   // 16 rows * 46 chunks
#define CTX_K      6     // ceil(736/128) chunks per thread

__device__ __forceinline__ unsigned short f2bf(float f) {
    unsigned int u = __float_as_uint(f);
    u += 0x7FFFu + ((u >> 16) & 1u);      // round-to-nearest-even
    return (unsigned short)(u >> 16);
}

__device__ __forceinline__ unsigned int pack2bf(float lo, float hi) {
    return (unsigned int)f2bf(lo) | ((unsigned int)f2bf(hi) << 16);
}

union Frag { uint4 q[2]; v16bf v; };

// A-fragment (16x32 bf16, MxK): lane holds row M=lane&15; lanes>=16 take the
// +8 K-offset half. Two contiguous 16-byte LDS loads per lane.
__device__ __forceinline__ v16bf load_a(const unsigned short* lds, int m,
                                        int stride, int kt, int laneHi) {
    const unsigned short* p = lds + m * stride + kt * 32 + laneHi * 8;
    Frag f;
    f.q[0] = *(const uint4*)(p);
    f.q[1] = *(const uint4*)(p + 16);
    return f.v;
}

// B-fragment from pre-swizzled weights: 32 bytes contiguous per lane per block.
__device__ __forceinline__ v16bf load_b(const unsigned short* w, int blk, int lane) {
    const unsigned short* p = w + ((size_t)blk * 32 + lane) * 16;
    Frag f;
    f.q[0] = *(const uint4*)(p);
    f.q[1] = *(const uint4*)(p + 8);
    return f.v;
}

// ---------------------------------------------------------------------------
// Prep: swizzle W[Kreal x N] (row-major f32) into bf16 WMMA B-fragment blocks.
// grid = numKT*numNT blocks of 32 threads; block b = kt*numNT + nt.
// ---------------------------------------------------------------------------
__global__ void swizzle_w(const float* __restrict__ src,
                          unsigned short* __restrict__ dst,
                          int Kreal, int N, int numNT) {
    const int blk    = blockIdx.x;
    const int kt     = blk / numNT;
    const int nt     = blk % numNT;
    const int lane   = threadIdx.x;
    const int laneHi = lane >> 4;
    const int n      = nt * 16 + (lane & 15);
    unsigned short* out = dst + ((size_t)blk * 32 + lane) * 16;
#pragma unroll
    for (int e = 0; e < 16; ++e) {
        const int k = kt * 32 + e + ((e >= 8) ? 8 : 0) + laneHi * 8;
        const float v = (k < Kreal) ? src[(size_t)k * N + n] : 0.0f;
        out[e] = f2bf(v);
    }
}

// ---------------------------------------------------------------------------
// Fused critic kernel: block = (16 batch rows) x (twin = blockIdx.y), 4 waves.
// ---------------------------------------------------------------------------
__global__ void __launch_bounds__(128)
critic_kernel(const float* __restrict__ obs, const float* __restrict__ act,
              const float* __restrict__ ctx,
              const unsigned short* __restrict__ w1a_s, const float* __restrict__ b1a,
              const unsigned short* __restrict__ w2a_s, const float* __restrict__ b2a,
              const unsigned short* __restrict__ w1b_s, const float* __restrict__ b1b,
              const unsigned short* __restrict__ w2b_s, const float* __restrict__ b2b,
              const unsigned short* __restrict__ r1a_s, const float* __restrict__ rb1a,
              const float* __restrict__ rw2a, const float* __restrict__ rb2a,
              const unsigned short* __restrict__ r1b_s, const float* __restrict__ rb1b,
              const float* __restrict__ rw2b, const float* __restrict__ rb2b,
              float* __restrict__ out) {
    __shared__ unsigned short xs[16 * XS_STRIDE];  // x tile  [16 x 288] bf16
    __shared__ unsigned short hs[16 * HS_STRIDE];  // h / p   [16 x <=256] bf16
    __shared__ float qacc[16];

    const int tid    = threadIdx.x;
    const int lane   = tid & 31;
    const int wave   = tid >> 5;
    const int laneHi = (lane >> 4) & 1;
    const int lanelo = lane & 15;
    const int b0     = blockIdx.x * 16;
    const int twin   = blockIdx.y;

    const unsigned short* w1 = twin ? w1b_s : w1a_s;
    const float*          b1 = twin ? b1b   : b1a;
    const unsigned short* w2 = twin ? w2b_s : w2a_s;
    const float*          b2 = twin ? b2b   : b2a;
    const unsigned short* r1 = twin ? r1b_s : r1a_s;
    const float*          rb1 = twin ? rb1b : rb1a;
    const float*          rw2 = twin ? rw2b : rw2a;
    const float*          rb2 = twin ? rb2b : rb2a;

    // body_act (invariant over objects): cols 0..63 obs[:, :64], 64..79 act.
    for (int i = tid; i < 16 * 80; i += 128) {
        const int r = i / 80, c = i % 80;
        const float v = (c < 64) ? obs[(size_t)(b0 + r) * 100 + c]
                                 : act[(size_t)(b0 + r) * 16 + (c - 64)];
        xs[r * XS_STRIDE + c] = f2bf(v);
    }
    // K padding 264..287 -> 0
    for (int i = tid; i < 16 * (K1_PAD - K1_REAL); i += 128) {
        const int r = i / 24, c = i % 24;
        xs[r * XS_STRIDE + K1_REAL + c] = 0;
    }

    // ---- ctx software pipeline (float4 granularity) ----
    // chunk i4 = tid + k*128 of 736 float4s; row r = i4/46, col4 c4 = i4%46.
    // Per-object advance = DIM_CTX floats (layout [B, NB_OBJ, DIM_CTX]).
    const float4* ctx_p4[CTX_K];
    int           xs_off4[CTX_K];           // element offset (multiple of 4)
#pragma unroll
    for (int k = 0; k < CTX_K; ++k) {
        const int i4 = tid + k * 128;
        const int r = i4 / CTX_F4, c4 = i4 % CTX_F4;
        ctx_p4[k]  = (const float4*)(ctx + (size_t)(b0 + r) * NB_OBJ * DIM_CTX
                                         + c4 * 4);
        xs_off4[k] = r * XS_STRIDE + 80 + c4 * 4;
    }
    const bool hasLast = (tid < CTX_CHUNKS - 5 * 128);   // tid < 96
    float4 creg[CTX_K];
#pragma unroll
    for (int k = 0; k < CTX_K - 1; ++k) creg[k] = ctx_p4[k][0];
    if (hasLast) creg[CTX_K - 1] = ctx_p4[CTX_K - 1][0];

    const v8f vzero = {0.f, 0.f, 0.f, 0.f, 0.f, 0.f, 0.f, 0.f};
    v8f pacc[3] = {vzero, vzero, vzero};   // pooled phi output (3 N-tiles/wave)

    for (int o = 0; o < NB_OBJ; ++o) {
        // commit prefetched ctx slice to LDS (packed bf16, 8B stores).
        // Safe: all xs readers of iter o-1 finished before the post-h barrier
        // of iter o-1, which every wave has passed before reaching here.
#pragma unroll
        for (int k = 0; k < CTX_K; ++k) {
            if (k == CTX_K - 1 && !hasLast) break;
            const float4 v = creg[k];
            uint2 pk;
            pk.x = pack2bf(v.x, v.y);
            pk.y = pack2bf(v.z, v.w);
            *(uint2*)(xs + xs_off4[k]) = pk;
        }
        __syncthreads();   // xs ready (also fences iter o-1 GEMM2 hs reads
                           // from this iteration's GEMM1 hs writes)
        // issue next object's loads; wait lands at next iteration's commit,
        // hidden behind the ~60 WMMAs below.
        if (o + 1 < NB_OBJ) {
            const size_t adv = (size_t)(o + 1) * CTX_F4;   // float4 stride/obj
#pragma unroll
            for (int k = 0; k < CTX_K - 1; ++k) creg[k] = ctx_p4[k][adv];
            if (hasLast) creg[CTX_K - 1] = ctx_p4[CTX_K - 1][adv];
        }

        // ---- GEMM1: [16,288] x [288,256], 4 N-tiles per wave ----
        v8f acc[4] = {vzero, vzero, vzero, vzero};
        for (int kt = 0; kt < 9; ++kt) {
            const v16bf a = load_a(xs, lanelo, XS_STRIDE, kt, laneHi);
#pragma unroll
            for (int j = 0; j < 4; ++j) {
                const int nt = wave * 4 + j;
                const v16bf bw = load_b(w1, kt * 16 + nt, lane);
                acc[j] = __builtin_amdgcn_wmma_f32_16x16x32_bf16(
                    false, a, false, bw, (short)0, acc[j], false, false);
            }
        }
        // bias + ReLU -> hs (bf16)
#pragma unroll
        for (int j = 0; j < 4; ++j) {
            const int n = (wave * 4 + j) * 16 + lanelo;
            const float bias = b1[n];
#pragma unroll
            for (int r = 0; r < 8; ++r) {
                const int m = r + laneHi * 8;
                hs[m * HS_STRIDE + n] = f2bf(fmaxf(acc[j][r] + bias, 0.f));
            }
        }
        __syncthreads();   // hs ready (also fences this iter's xs reads from
                           // next iteration's ctx commit)

        // ---- GEMM2: [16,256] x [256,192], 3 N-tiles per wave ----
        v8f acc2[3] = {vzero, vzero, vzero};
        for (int kt = 0; kt < 8; ++kt) {
            const v16bf a = load_a(hs, lanelo, HS_STRIDE, kt, laneHi);
#pragma unroll
            for (int j = 0; j < 3; ++j) {
                const int nt = wave * 3 + j;
                const v16bf bw = load_b(w2, kt * 12 + nt, lane);
                acc2[j] = __builtin_amdgcn_wmma_f32_16x16x32_bf16(
                    false, a, false, bw, (short)0, acc2[j], false, false);
            }
        }
        // bias + ReLU, accumulate object sum-pool in registers
#pragma unroll
        for (int j = 0; j < 3; ++j) {
            const int n = (wave * 3 + j) * 16 + lanelo;
            const float bias = b2[n];
#pragma unroll
            for (int r = 0; r < 8; ++r)
                pacc[j][r] += fmaxf(acc2[j][r] + bias, 0.f);
        }
    }

    __syncthreads();    // all GEMM2 reads of hs done; reuse hs for pooled p
#pragma unroll
    for (int j = 0; j < 3; ++j) {
        const int n = (wave * 3 + j) * 16 + lanelo;
#pragma unroll
        for (int r = 0; r < 8; ++r) {
            const int m = r + laneHi * 8;
            hs[m * HS_STRIDE + n] = f2bf(pacc[j][r]);
        }
    }
    if (tid < 16) qacc[tid] = 0.f;
    __syncthreads();

    // ---- rho: relu([16,192] x [192,256] + b) . w2 + b2 ----
    v8f accr[4] = {vzero, vzero, vzero, vzero};
    for (int kt = 0; kt < 6; ++kt) {
        const v16bf a = load_a(hs, lanelo, HS_STRIDE, kt, laneHi);
#pragma unroll
        for (int j = 0; j < 4; ++j) {
            const int nt = wave * 4 + j;
            const v16bf bw = load_b(r1, kt * 16 + nt, lane);
            accr[j] = __builtin_amdgcn_wmma_f32_16x16x32_bf16(
                false, a, false, bw, (short)0, accr[j], false, false);
        }
    }
    float rowsum[8] = {0.f, 0.f, 0.f, 0.f, 0.f, 0.f, 0.f, 0.f};
#pragma unroll
    for (int j = 0; j < 4; ++j) {
        const int n = (wave * 4 + j) * 16 + lanelo;
        const float bias = rb1[n];
        const float w2v  = rw2[n];
#pragma unroll
        for (int r = 0; r < 8; ++r)
            rowsum[r] += fmaxf(accr[j][r] + bias, 0.f) * w2v;
    }
#pragma unroll
    for (int r = 0; r < 8; ++r)
        atomicAdd(&qacc[r + laneHi * 8], rowsum[r]);
    __syncthreads();

    if (tid < 16)
        out[(size_t)twin * BATCH + b0 + tid] = qacc[tid] + rb2[0];
}

// ---------------------------------------------------------------------------
// Workspace layout (bf16 elements):
//   w1 swizzled : 9*16*512 = 73728  per twin
//   w2 swizzled : 8*12*512 = 49152  per twin
//   r1 swizzled : 6*16*512 = 49152  per twin
// total = 344064 elems = 688128 bytes.
// ---------------------------------------------------------------------------
extern "C" void kernel_launch(void* const* d_in, const int* in_sizes, int n_in,
                              void* d_out, int out_size, void* d_ws, size_t ws_size,
                              hipStream_t stream) {
    const float* obs     = (const float*)d_in[0];
    const float* act     = (const float*)d_in[1];
    const float* ctx     = (const float*)d_in[2];
    const float* phi_w1a = (const float*)d_in[3];
    const float* phi_b1a = (const float*)d_in[4];
    const float* phi_w2a = (const float*)d_in[5];
    const float* phi_b2a = (const float*)d_in[6];
    const float* phi_w1b = (const float*)d_in[7];
    const float* phi_b1b = (const float*)d_in[8];
    const float* phi_w2b = (const float*)d_in[9];
    const float* phi_b2b = (const float*)d_in[10];
    const float* rho_w1a = (const float*)d_in[11];
    const float* rho_b1a = (const float*)d_in[12];
    const float* rho_w2a = (const float*)d_in[13];
    const float* rho_b2a = (const float*)d_in[14];
    const float* rho_w1b = (const float*)d_in[15];
    const float* rho_b1b = (const float*)d_in[16];
    const float* rho_w2b = (const float*)d_in[17];
    const float* rho_b2b = (const float*)d_in[18];

    unsigned short* ws   = (unsigned short*)d_ws;
    unsigned short* w1a_s = ws;
    unsigned short* w2a_s = w1a_s + 73728;
    unsigned short* r1a_s = w2a_s + 49152;
    unsigned short* w1b_s = r1a_s + 49152;
    unsigned short* w2b_s = w1b_s + 73728;
    unsigned short* r1b_s = w2b_s + 49152;

    // weight swizzle (cheap; runs every call, deterministic)
    swizzle_w<<<dim3(9 * 16), dim3(32), 0, stream>>>(phi_w1a, w1a_s, K1_REAL, N1, 16);
    swizzle_w<<<dim3(8 * 12), dim3(32), 0, stream>>>(phi_w2a, w2a_s, K2, N2, 12);
    swizzle_w<<<dim3(6 * 16), dim3(32), 0, stream>>>(rho_w1a, r1a_s, KR, NR, 16);
    swizzle_w<<<dim3(9 * 16), dim3(32), 0, stream>>>(phi_w1b, w1b_s, K1_REAL, N1, 16);
    swizzle_w<<<dim3(8 * 12), dim3(32), 0, stream>>>(phi_w2b, w2b_s, K2, N2, 12);
    swizzle_w<<<dim3(6 * 16), dim3(32), 0, stream>>>(rho_w1b, r1b_s, KR, NR, 16);

    dim3 grid(BATCH / 16, 2);
    critic_kernel<<<grid, dim3(128), 0, stream>>>(
        obs, act, ctx,
        w1a_s, phi_b1a, w2a_s, phi_b2a,
        w1b_s, phi_b1b, w2b_s, phi_b2b,
        r1a_s, rho_b1a, rho_w2a, rho_b2a,
        r1b_s, rho_b1b, rho_w2b, rho_b2b,
        (float*)d_out);
}
